// MAF_42417097016686
// MI455X (gfx1250) — matrix operand order repaired and checked by reading, not compiled, not run on hardware
//
#include <hip/hip_runtime.h>
#include <hip/hip_bf16.h>

// ---------------------------------------------------------------------------
// MAF forward, fused:  G = x @ W1m (bf16 WMMA, f32 accum)  ->  tanh -> 8x8 ->
// tanh -> 8x2 -> z = (x - mu) * exp(-alpha), log_det = -sum(alpha)
// ---------------------------------------------------------------------------

typedef float  v8f   __attribute__((ext_vector_type(8)));
typedef __bf16 v16bf __attribute__((ext_vector_type(16)));

#define B_ROWS   16384
#define D_DIM    256
#define L_LAYERS 255
#define H_DIM    8

// block geometry
#define BDIM        256   // 8 waves
#define LPB         8     // layers per block (N-chunk = 64 cols = 4 WMMA N-tiles)
#define ROWS_PER_IT 128   // 8 waves x 16 rows
#define ITERS       8     // 1024 rows per block

__device__ __forceinline__ float fast_tanh(float v) {
    // tanh(v) = 1 - 2/(exp(2v)+1); saturates correctly at +/-1 for large |v|
    float e = __expf(2.0f * v);
    return 1.0f - 2.0f / (e + 1.0f);
}

// pack 4 float4 (one contiguous 16-wide K run) into the 16xbf16 WMMA A fragment
__device__ __forceinline__ v16bf cvtA(const float4& a0, const float4& a1,
                                      const float4& a2, const float4& a3) {
    v16bf A;
    A[0]  = (__bf16)a0.x; A[1]  = (__bf16)a0.y; A[2]  = (__bf16)a0.z; A[3]  = (__bf16)a0.w;
    A[4]  = (__bf16)a1.x; A[5]  = (__bf16)a1.y; A[6]  = (__bf16)a1.z; A[7]  = (__bf16)a1.w;
    A[8]  = (__bf16)a2.x; A[9]  = (__bf16)a2.y; A[10] = (__bf16)a2.z; A[11] = (__bf16)a2.w;
    A[12] = (__bf16)a3.x; A[13] = (__bf16)a3.y; A[14] = (__bf16)a3.z; A[15] = (__bf16)a3.w;
    return A;
}

// ---------------------------------------------------------------------------
// Kernel 1: seed column 0 of the flow (initial_param) and log_det = -alpha0.
// Output layout: d_out = [ z_rev (B*D floats) | log_det (B floats) ]
// z_rev[m][255] corresponds to original column 0.
// ---------------------------------------------------------------------------
__global__ __launch_bounds__(256) void maf_init(const float* __restrict__ x,
                                                const float* __restrict__ ip,
                                                float* __restrict__ zout,
                                                float* __restrict__ logdet) {
    int m = blockIdx.x * blockDim.x + threadIdx.x;
    if (m < B_ROWS) {
        float mu0 = ip[0];
        float a0  = ip[1];
        zout[(size_t)m * D_DIM + (D_DIM - 1)] = (x[(size_t)m * D_DIM] - mu0) * __expf(-a0);
        logdet[m] = -a0;
    }
}

// ---------------------------------------------------------------------------
// Kernel 2: main fused kernel.
// grid = (32 layer-chunks, 16 row-chunks), block = 256 threads = 8 waves.
// ---------------------------------------------------------------------------
__global__ __launch_bounds__(256) void maf_main(const float* __restrict__ x,
                                                const float* __restrict__ W1,
                                                const float* __restrict__ b1,
                                                const float* __restrict__ W2,
                                                const float* __restrict__ b2,
                                                const float* __restrict__ W3,
                                                const float* __restrict__ b3,
                                                float* __restrict__ zout,
                                                float* __restrict__ logdet) {
    // LDS: causal-masked W1 slice, bf16, transposed to [n][d] so the WMMA
    // B-operand is a contiguous 32B run per lane (2x ds_load_b128).
    __shared__ __align__(32) __bf16 wtT[LPB * H_DIM][D_DIM];   // 32 KB
    __shared__ float Gs[8][16][16];                            // per-wave C-tile scratch, 8 KB
    __shared__ float b1s[LPB][H_DIM];
    __shared__ float W2s[LPB][H_DIM][H_DIM];
    __shared__ float b2s[LPB][H_DIM];
    __shared__ float W3s[LPB][H_DIM][2];
    __shared__ float b3s[LPB][2];

    const int nb   = blockIdx.x;          // layer chunk: layers [nb*8, nb*8+8)
    const int mb   = blockIdx.y;          // row chunk:   rows   [mb*1024, +1024)
    const int tid  = threadIdx.x;
    const int wave = tid >> 5;
    const int lane = tid & 31;
    const int hl   = lane >> 4;           // wave half (0/1)
    const int ml   = lane & 15;           // M index within tile / N index for B
    const int l0   = nb * LPB;

    // ---- pack W1 tile (apply causal mask; zero the nonexistent layer 255) ----
    for (int i = tid; i < LPB * H_DIM * D_DIM; i += BDIM) {
        int n = i >> 8;                   // 0..63
        int d = i & (D_DIM - 1);
        int ll = l0 + (n >> 3);
        int h  = n & 7;
        float w = 0.0f;
        if (ll < L_LAYERS && d <= ll) w = W1[(size_t)ll * (D_DIM * H_DIM) + d * H_DIM + h];
        wtT[n][d] = (__bf16)w;
    }
    // ---- pack tail parameters ----
    for (int i = tid; i < LPB * H_DIM; i += BDIM) {
        int lloc = i >> 3, h = i & 7;
        int ll = l0 + lloc;
        bool v = (ll < L_LAYERS);
        b1s[lloc][h]    = v ? b1[ll * H_DIM + h]     : 0.0f;
        b2s[lloc][h]    = v ? b2[ll * H_DIM + h]     : 0.0f;
        W3s[lloc][h][0] = v ? W3[ll * 16 + h * 2 + 0] : 0.0f;
        W3s[lloc][h][1] = v ? W3[ll * 16 + h * 2 + 1] : 0.0f;
        if (h < 2) b3s[lloc][h] = v ? b3[ll * 2 + h] : 0.0f;
    }
    for (int i = tid; i < LPB * 64; i += BDIM) {
        int lloc = i >> 6, hk = i & 63;
        int ll = l0 + lloc;
        W2s[lloc][hk >> 3][hk & 7] = (ll < L_LAYERS) ? W2[ll * 64 + hk] : 0.0f;
    }
    __syncthreads();

    const int loff = hl * 16;    // A: lanes0-15 hold K 0..15, lanes16-31 hold K 16..31
    const int koff = hl * 16;    // B: lanes0-15 hold K 0..15, lanes16-31 hold K 16..31

    for (int it = 0; it < ITERS; ++it) {
        const int rowbase = mb * 1024 + it * ROWS_PER_IT + wave * 16;
        const int row     = rowbase + ml;
        const float* xr   = x + (size_t)row * D_DIM;

        if (it + 1 < ITERS) {
            // warm L2 for the next row tile (global_prefetch_b8)
            __builtin_prefetch(xr + (size_t)ROWS_PER_IT * D_DIM, 0, 1);
        }

        v8f C[4];
        C[0] = v8f{}; C[1] = v8f{}; C[2] = v8f{}; C[3] = v8f{};

        // ---- GEMM: K = 256 in 8 steps of 32, N = 64 as 4 WMMA tiles.
        // A is software-pipelined 2 steps deep (global loads) with a 3-deep
        // rotating bf16 fragment buffer so the cvt for step kk+1 never writes
        // registers read by the immediately preceding WMMA group (no hazard
        // NOPs). All 8 B ds_loads are batched ahead of the 4 WMMAs so the
        // dscnt waits stagger instead of draining to zero each time.
        v16bf  A[3];
        float4 f[2][4];
        // prologue: K-steps 0 and 1 in flight, fragment 0 ready
        #pragma unroll
        for (int p = 0; p < 2; ++p) {
            const int k0 = p * 32;
            f[p][0] = *reinterpret_cast<const float4*>(xr + k0 + loff);
            f[p][1] = *reinterpret_cast<const float4*>(xr + k0 + loff + 4);
            f[p][2] = *reinterpret_cast<const float4*>(xr + k0 + loff + 8);
            f[p][3] = *reinterpret_cast<const float4*>(xr + k0 + loff + 12);
        }
        A[0] = cvtA(f[0][0], f[0][1], f[0][2], f[0][3]);

        #pragma unroll
        for (int kk = 0; kk < 8; ++kk) {
            const int k0 = kk * 32;
            if (kk < 6) {                 // issue K-step kk+2 global loads
                const int kn = k0 + 64;
                f[kk & 1][0] = *reinterpret_cast<const float4*>(xr + kn + loff);
                f[kk & 1][1] = *reinterpret_cast<const float4*>(xr + kn + loff + 4);
                f[kk & 1][2] = *reinterpret_cast<const float4*>(xr + kn + loff + 8);
                f[kk & 1][3] = *reinterpret_cast<const float4*>(xr + kn + loff + 12);
            }
            // batch all B-tile LDS loads for this K-step
            v16bf B0 = *reinterpret_cast<const v16bf*>(&wtT[ 0 + ml][k0 + koff]);
            v16bf B1 = *reinterpret_cast<const v16bf*>(&wtT[16 + ml][k0 + koff]);
            v16bf B2 = *reinterpret_cast<const v16bf*>(&wtT[32 + ml][k0 + koff]);
            v16bf B3 = *reinterpret_cast<const v16bf*>(&wtT[48 + ml][k0 + koff]);
            if (kk < 7) {                 // convert K-step kk+1 (loaded last step)
                const int q = (kk + 1) & 1;
                A[(kk + 1) % 3] = cvtA(f[q][0], f[q][1], f[q][2], f[q][3]);
            }
            const v16bf& Ac = A[kk % 3];
            C[0] = __builtin_amdgcn_wmma_f32_16x16x32_bf16(false, Ac, false, B0, (short)0, C[0], false, false);
            C[1] = __builtin_amdgcn_wmma_f32_16x16x32_bf16(false, Ac, false, B1, (short)0, C[1], false, false);
            C[2] = __builtin_amdgcn_wmma_f32_16x16x32_bf16(false, Ac, false, B2, (short)0, C[2], false, false);
            C[3] = __builtin_amdgcn_wmma_f32_16x16x32_bf16(false, Ac, false, B3, (short)0, C[3], false, false);
        }

        // ---- tails: per N-tile, redistribute C through LDS, run tiny MLP ----
        float asum = 0.0f;
        #pragma unroll
        for (int t = 0; t < 4; ++t) {
            // C layout: VGPR r -> M = r + hl*8, N = lane&15
            #pragma unroll
            for (int r = 0; r < 8; ++r)
                Gs[wave][r + hl * 8][ml] = C[t][r];
            asm volatile("s_wait_dscnt 0" ::: "memory");

            const int lloc = t * 2 + hl;        // this lane's layer within chunk
            const int ll   = l0 + lloc;

            float h1[8];
            #pragma unroll
            for (int h = 0; h < 8; ++h)
                h1[h] = fast_tanh(Gs[wave][ml][hl * 8 + h] + b1s[lloc][h]);

            float h2[8];
            #pragma unroll
            for (int k = 0; k < 8; ++k) {
                float acc = b2s[lloc][k];
                #pragma unroll
                for (int h = 0; h < 8; ++h) acc += h1[h] * W2s[lloc][h][k];
                h2[k] = fast_tanh(acc);
            }
            float mu = b3s[lloc][0];
            float al = b3s[lloc][1];
            #pragma unroll
            for (int h = 0; h < 8; ++h) {
                mu += h2[h] * W3s[lloc][h][0];
                al += h2[h] * W3s[lloc][h][1];
            }

            if (ll < L_LAYERS) {
                // layer ll predicts original column (ll+1); reversed index = 254-ll
                float xv = x[(size_t)row * D_DIM + ll + 1];
                zout[(size_t)row * D_DIM + (D_DIM - 2 - ll)] = (xv - mu) * __expf(-al);
                asum += al;
            }
            asm volatile("" ::: "memory");  // keep Gs reads of tile t ahead of tile t+1 stores
        }
        atomicAdd(&logdet[row], -asum);
    }
}

// ---------------------------------------------------------------------------
// Host-side launcher
// ---------------------------------------------------------------------------
extern "C" void kernel_launch(void* const* d_in, const int* in_sizes, int n_in,
                              void* d_out, int out_size, void* d_ws, size_t ws_size,
                              hipStream_t stream) {
    (void)in_sizes; (void)n_in; (void)out_size; (void)d_ws; (void)ws_size;
    const float* x  = (const float*)d_in[0];
    const float* ip = (const float*)d_in[1];
    const float* W1 = (const float*)d_in[2];
    const float* b1 = (const float*)d_in[3];
    const float* W2 = (const float*)d_in[4];
    const float* b2 = (const float*)d_in[5];
    const float* W3 = (const float*)d_in[6];
    const float* b3 = (const float*)d_in[7];

    float* zout   = (float*)d_out;
    float* logdet = zout + (size_t)B_ROWS * D_DIM;

    maf_init<<<B_ROWS / 256, 256, 0, stream>>>(x, ip, zout, logdet);
    maf_main<<<dim3(32, 16), 256, 0, stream>>>(x, W1, b1, W2, b2, W3, b3, zout, logdet);
}
